// HierarchicalOTAlignment_31353261261218
// MI455X (gfx1250) — matrix operand order repaired and checked by reading, not compiled, not run on hardware
//
#include <hip/hip_runtime.h>
#include <hip/hip_bf16.h>

typedef _Float16 v8h  __attribute__((ext_vector_type(8)));
typedef _Float16 v16h __attribute__((ext_vector_type(16)));
typedef float    v8f  __attribute__((ext_vector_type(8)));

union H16 { v16h v; v8h h[2]; };

#define BATCH 16
#define DIM   2048           // M == N == 2048
#define EPS   0.1f
#define NITER 100
#define SK    16.0f          // K matrix scale (keeps fp16 out of subnormals)
#define SX    262144.0f      // 2^18 scale for x before fp16 (exact power of two)
#define KWAVES 8             // waves per block, each owns a DIM/KWAVES K-slice
#define KSLICE (DIM / KWAVES)

// ---------------------------------------------------------------------------
// Prep: K_h[b,i,j] = fp16(SK * exp(-C/eps)), KT_h[b,j,i] = same (transposed).
// 32x32 tiles via LDS for coalesced transposed stores.
// ---------------------------------------------------------------------------
__global__ __launch_bounds__(256) void prep_kernel(const float* __restrict__ C,
                                                   _Float16* __restrict__ Kh,
                                                   _Float16* __restrict__ KT) {
  __shared__ _Float16 tile[32][33];
  const int b  = blockIdx.z;
  const int i0 = blockIdx.y * 32;
  const int j0 = blockIdx.x * 32;
  const int tx = threadIdx.x;       // 0..31
  const int ty = threadIdx.y;       // 0..7
  const size_t base = (size_t)b * DIM * DIM;
#pragma unroll
  for (int r = 0; r < 4; ++r) {
    const int i = i0 + ty + 8 * r;
    const float c = C[base + (size_t)i * DIM + j0 + tx];
    const _Float16 k = (_Float16)(SK * expf(c * (-1.0f / EPS)));
    Kh[base + (size_t)i * DIM + j0 + tx] = k;
    tile[ty + 8 * r][tx] = k;
  }
  __syncthreads();
#pragma unroll
  for (int r = 0; r < 4; ++r) {
    const int j = j0 + ty + 8 * r;
    KT[base + (size_t)j * DIM + i0 + tx] = tile[tx][ty + 8 * r];
  }
}

// ---------------------------------------------------------------------------
// Init: y_h = 1.0 (v0 = 0 -> exp(v) = 1)
// ---------------------------------------------------------------------------
__global__ void init_kernel(_Float16* __restrict__ y_h) {
  const int t = blockIdx.x * 256 + threadIdx.x;
  if (t < BATCH * DIM) y_h[t] = (_Float16)1.0f;
}

// ---------------------------------------------------------------------------
// One Sinkhorn half-step as a WMMA GEMV:
//   r[i] = sum_j Kmat[i,j] * vin[j]   (Kmat is fp16, scaled by SK)
//   vout[i]   = marg[i] * numer_scale / r[i]
//   vout_h[i] = fp16(vout[i] * out_scale)
//
// One 256-thread block per 16-row tile; its 8 waves split the K dimension
// (256 K-values each, 8 WMMAs per wave) and combine partials through LDS.
// B operand = vin replicated over all 16 columns, so all D-columns are equal.
// A fragment per 16-bit 16x32 layout: lane<16 -> row=lane, K 0-7 & 16-23;
//                                     lane>=16 -> row=lane-16, K 8-15 & 24-31.
// B fragment: same K layout as A (lane<16 -> K 0-7 & 16-23; lane>=16 -> K 8-15 & 24-31).
// ---------------------------------------------------------------------------
__global__ __launch_bounds__(256) void gemv16_kernel(const _Float16* __restrict__ Kmat,
                                                     const _Float16* __restrict__ vin,
                                                     const float* __restrict__ marg,
                                                     float* __restrict__ vout,
                                                     _Float16* __restrict__ vout_h,
                                                     float numer_scale,
                                                     float out_scale) {
  __shared__ float sdata[KWAVES][16];

  const int batch = blockIdx.x >> 7;      // DIM/16 = 128 row-blocks per batch
  const int rblk  = blockIdx.x & 127;
  const int w     = threadIdx.x >> 5;     // wave in block: 0..7 (K slice id)
  const int lane  = threadIdx.x & 31;
  const int r  = lane & 15;
  const int hs = lane >> 4;               // half-wave select (0/1)

  // Pre-offset base pointers by this wave's K-slice start so the loop below
  // has compile-time-constant, wave-uniform bounds (scalar loop control).
  const _Float16* __restrict__ arow =
      Kmat + (size_t)batch * DIM * DIM + (size_t)(rblk * 16 + r) * DIM +
      w * KSLICE + hs * 8;
  const _Float16* __restrict__ yb = vin + batch * DIM + w * KSLICE + hs * 8;

  v8f acc = {0.f, 0.f, 0.f, 0.f, 0.f, 0.f, 0.f, 0.f};
#pragma unroll 4
  for (int t = 0; t < KSLICE; t += 32) {
    H16 A, B;
    A.h[0] = *(const v8h*)(arow + t);        // K = t+hs*8 .. +7
    A.h[1] = *(const v8h*)(arow + t + 16);   // K = t+16+hs*8 .. +7
    B.h[0] = *(const v8h*)(yb + t);          // K = t+hs*8 .. +7
    B.h[1] = *(const v8h*)(yb + t + 16);     // K = t+16+hs*8 .. +7
    acc = __builtin_amdgcn_wmma_f32_16x16x32_f16(
        false, A.v, false, B.v, (short)0, acc, false, false);
  }

  // All 16 D-columns are identical. VGPR g of lanes 0-15 holds row g,
  // VGPR g of lanes 16-31 holds row 8+g. Lanes 0/16 stage partials to LDS.
  if (r == 0) {
#pragma unroll
    for (int g = 0; g < 8; ++g) sdata[w][hs * 8 + g] = acc[g];
  }
  __syncthreads();

  // Threads 0..15: sum the 8 K-slice partials for one row, finish the update.
  if (threadIdx.x < 16) {
    const int row = rblk * 16 + threadIdx.x;
    float s = 0.0f;
#pragma unroll
    for (int ww = 0; ww < KWAVES; ++ww) s += sdata[ww][threadIdx.x];
    const float val = marg[batch * DIM + row] * numer_scale / s;
    vout[batch * DIM + row]   = val;
    vout_h[batch * DIM + row] = (_Float16)(val * out_scale);
  }
}

// ---------------------------------------------------------------------------
// Finalize: P = x * exp(-C/eps) * y (fp32, from original C); per-block
// partial sums of P*C for the distance (deterministic two-stage reduction).
// ---------------------------------------------------------------------------
__global__ __launch_bounds__(256) void final_kernel(const float* __restrict__ C,
                                                    const float* __restrict__ x,
                                                    const float* __restrict__ y,
                                                    float* __restrict__ P,
                                                    float* __restrict__ partials) {
  __shared__ float sdata[256];
  const int b = blockIdx.y;
  const size_t base = (size_t)b * DIM * DIM;
  const size_t flat = ((size_t)blockIdx.x * 256 + threadIdx.x) * 4;
  const int i = (int)(flat >> 11);
  const int j = (int)(flat & (DIM - 1));

  const float4 c = *(const float4*)(C + base + flat);
  const float xi = x[b * DIM + i];
  const float* yv = y + b * DIM + j;

  float4 p;
  p.x = xi * expf(c.x * (-1.0f / EPS)) * yv[0];
  p.y = xi * expf(c.y * (-1.0f / EPS)) * yv[1];
  p.z = xi * expf(c.z * (-1.0f / EPS)) * yv[2];
  p.w = xi * expf(c.w * (-1.0f / EPS)) * yv[3];
  *(float4*)(P + base + flat) = p;

  float local = p.x * c.x + p.y * c.y + p.z * c.z + p.w * c.w;
  sdata[threadIdx.x] = local;
  __syncthreads();
#pragma unroll
  for (int s = 128; s > 0; s >>= 1) {
    if (threadIdx.x < s) sdata[threadIdx.x] += sdata[threadIdx.x + s];
    __syncthreads();
  }
  if (threadIdx.x == 0) partials[b * 4096 + blockIdx.x] = sdata[0];
}

__global__ __launch_bounds__(256) void reduce_kernel(const float* __restrict__ partials,
                                                     float* __restrict__ dist) {
  __shared__ float sdata[256];
  const int b = blockIdx.x;
  float acc = 0.0f;
  for (int k = threadIdx.x; k < 4096; k += 256) acc += partials[b * 4096 + k];
  sdata[threadIdx.x] = acc;
  __syncthreads();
#pragma unroll
  for (int s = 128; s > 0; s >>= 1) {
    if (threadIdx.x < s) sdata[threadIdx.x] += sdata[threadIdx.x + s];
    __syncthreads();
  }
  if (threadIdx.x == 0) dist[b] = sdata[0];
}

// ---------------------------------------------------------------------------
extern "C" void kernel_launch(void* const* d_in, const int* in_sizes, int n_in,
                              void* d_out, int out_size, void* d_ws, size_t ws_size,
                              hipStream_t stream) {
  (void)in_sizes; (void)n_in; (void)out_size; (void)ws_size;

  const float* C  = (const float*)d_in[0];   // (16, 2048, 2048)
  const float* a  = (const float*)d_in[1];   // (16, 2048)
  const float* bm = (const float*)d_in[2];   // (16, 2048)

  float* dist = (float*)d_out;               // (16,)
  float* Pout = dist + BATCH;                // (16, 2048, 2048) fp32

  // The 256 MB P region doubles as fp16 K (row-major) + K^T scratch during
  // iterations; final_kernel overwrites it with P (reads only C/x/y).
  _Float16* Kh = (_Float16*)Pout;
  _Float16* KT = Kh + (size_t)BATCH * DIM * DIM;

  // Workspace: vectors + distance partials (~0.7 MB).
  float*    x_f = (float*)d_ws;
  float*    y_f = x_f + BATCH * DIM;
  _Float16* x_h = (_Float16*)(y_f + BATCH * DIM);
  _Float16* y_h = x_h + BATCH * DIM;
  float*    part = (float*)(y_h + BATCH * DIM);   // 16 * 4096 floats

  // 1) Build fp16 K and K^T from C.
  prep_kernel<<<dim3(DIM / 32, DIM / 32, BATCH), dim3(32, 8), 0, stream>>>(C, Kh, KT);

  // 2) y = exp(v0) = 1.
  init_kernel<<<(BATCH * DIM + 255) / 256, 256, 0, stream>>>(y_h);

  // 3) 100 Sinkhorn iterations: x = a/(K y), y = b/(K^T x).
  //    Row pass:  r = SK*(K y)           -> x = a*SK/r,       x_h = x*SX
  //    Col pass:  z = SK*SX*(K^T x)      -> y = b*SK*SX/z,    y_h = y
  const int nblocks = BATCH * (DIM / 16);   // one block per 16-row tile: 2048
  for (int it = 0; it < NITER; ++it) {
    gemv16_kernel<<<nblocks, 256, 0, stream>>>(Kh, y_h, a,  x_f, x_h, SK,      SX);
    gemv16_kernel<<<nblocks, 256, 0, stream>>>(KT, x_h, bm, y_f, y_h, SK * SX, 1.0f);
  }

  // 4) P = x * exp(-C/eps) * y ; distance = sum(P*C) (deterministic reduce).
  final_kernel<<<dim3((DIM * DIM) / (256 * 4), BATCH), 256, 0, stream>>>(C, x_f, y_f, Pout, part);
  reduce_kernel<<<BATCH, 256, 0, stream>>>(part, dist);
}
